// CropConv_58454504898571
// MI455X (gfx1250) — matrix operand order repaired and checked by reading, not compiled, not run on hardware
//
#include <hip/hip_runtime.h>

typedef __bf16 bf16_t;
typedef __attribute__((ext_vector_type(8)))  __bf16 v8bf;
typedef __attribute__((ext_vector_type(16))) __bf16 v16bf;
typedef __attribute__((ext_vector_type(8)))  float  v8f;

#define IN_C   64
#define OUT_C  64
#define HH     128
#define WWID   128
#define IC_PAD 72                       // 144B pitch: 16B-aligned, conflict-free b128
#define P_COLS 34                       // 32-pixel strip + 1-col halo each side
#define P_ROWS 6                        // 4 output rows + 1-row halo each side
#define PATCH_ELEMS (P_ROWS * P_COLS * IC_PAD)   // 14688 bf16 per split (~29.4 KB)

#define NFRAG     (9 * 2 * 4)           // taps * ksteps * oc-tiles = 72 A-fragments
#define FRAG_VALS (NFRAG * 32 * 16)     // bf16 values per precision = 36864

static __device__ __forceinline__ v16bf cat16(v8bf a, v8bf b) {
    return __builtin_shufflevector(a, b, 0,1,2,3,4,5,6,7,8,9,10,11,12,13,14,15);
}

// ---- one-shot: pack bf16 hi/lo weight splits into WMMA A-fragment order ----
// ws layout: [rs][kstep][ocTile][lane][16 bf16]   (16x32 A, contiguous k = 16*half + j)
__global__ __launch_bounds__(256)
void pack_w_bf16x2(const float* __restrict__ w,
                   bf16_t* __restrict__ ws_hi,
                   bf16_t* __restrict__ ws_lo)
{
    int t = blockIdx.x * 256 + threadIdx.x;
    if (t >= FRAG_VALS) return;
    int j    = t & 15;                  // value index within lane
    int lane = (t >> 4) & 31;
    int frag = t >> 9;
    int ocT  = frag & 3;
    int kst  = (frag >> 2) & 1;
    int rs   = frag >> 3;               // 0..8
    int half = lane >> 4;
    int l    = lane & 15;
    int oc   = ocT * 16 + l;            // A-matrix M = lane%16
    // value j of lane-half h holds K = 16*half + j (contiguous, matches B)
    int K    = (half << 4) + j;
    int ic   = (kst << 5) + K;
    float v  = w[(oc * IN_C + ic) * 9 + rs];
    bf16_t hi = (bf16_t)v;
    bf16_t lo = (bf16_t)(v - (float)hi);
    ws_hi[t] = hi;
    ws_lo[t] = lo;
}

// ---- implicit-GEMM 3x3 conv, 3x bf16 split on V_WMMA_F32_16X16X32_BF16 ----
// block = (b, 4 output rows, 32-pixel strip); 8 waves = 4 oc-tiles x 2 strips,
// each wave keeps 4 row-accumulators and reuses every A-fragment across 4 rows.
__global__ __launch_bounds__(256)
void cropconv_wmma_bf16x3(const float* __restrict__ x,
                          const bf16_t* __restrict__ ws_hi,
                          const bf16_t* __restrict__ ws_lo,
                          float* __restrict__ out)
{
    __shared__ __align__(16) bf16_t ph[PATCH_ELEMS];   // hi split, [row][col][ic]
    __shared__ __align__(16) bf16_t pl[PATCH_ELEMS];   // lo split

    const int blk   = blockIdx.x;
    const int strip = blk & 3;             // 4 strips of 32 pixels across W
    const int hq    = (blk >> 2) & 31;     // 32 groups of 4 rows
    const int b     = blk >> 7;
    const int h0    = hq << 2;
    const int w0    = strip * 32;
    const int tid   = threadIdx.x;

    // stage zero-padded 6-row patch, splitting fp32 -> bf16 hi + residual lo
    for (int i = tid; i < IN_C * P_ROWS * P_COLS; i += 256) {
        int ic  = i / (P_ROWS * P_COLS);
        int rem = i - ic * (P_ROWS * P_COLS);
        int row = rem / P_COLS;
        int col = rem - row * P_COLS;
        int hh  = h0 - 1 + row;
        int ww  = w0 - 1 + col;
        float v = 0.0f;
        if ((unsigned)hh < 128u && (unsigned)ww < 128u)
            v = x[((b * IN_C + ic) * HH + hh) * WWID + ww];
        bf16_t hi = (bf16_t)v;
        bf16_t lo = (bf16_t)(v - (float)hi);
        int o = (row * P_COLS + col) * IC_PAD + ic;
        ph[o] = hi;
        pl[o] = lo;
    }
    __syncthreads();

    const int wid  = tid >> 5;
    const int lane = tid & 31;
    const int half = lane >> 4;
    const int l    = lane & 15;
    const int ocT  = wid & 3;
    const int oc0  = ocT * 16;
    const int wsub = (wid >> 2) * 16;
    const int wp   = w0 + wsub + l;

    // whole 4-row x 16-col tile inside crop window: zeros, skip all compute
    const bool colsMasked = ((w0 + wsub) >= 44) && ((w0 + wsub + 16) <= 84);
    if (colsMasked && (h0 >= 44) && ((h0 + 4) <= 84)) {
        for (int row = 0; row < 4; ++row)
            for (int v = 0; v < 8; ++v) {
                int oc = oc0 + half * 8 + v;
                out[((b * OUT_C + oc) * HH + (h0 + row)) * WWID + wp] = 0.0f;
            }
        return;
    }

    v8f acc[4];
#pragma unroll
    for (int row = 0; row < 4; ++row)
        acc[row] = (v8f){0.f, 0.f, 0.f, 0.f, 0.f, 0.f, 0.f, 0.f};

    const int icStart = half << 4;      // B layout: lane-half selects K+16

    for (int r = 0; r < 3; ++r) {
        for (int s = 0; s < 3; ++s) {
            const int rs      = r * 3 + s;
            const int basecol = wsub + l + s;
#pragma unroll
            for (int kst = 0; kst < 2; ++kst) {
                const int fbase = (((rs * 2 + kst) * 4 + ocT) * 32 + lane) * 16;
                const v8bf* ah = (const v8bf*)(ws_hi + fbase);
                const v8bf* al = (const v8bf*)(ws_lo + fbase);
                v16bf Ah = cat16(ah[0], ah[1]);
                v16bf Al = cat16(al[0], al[1]);
                const int boff = kst * 32 + icStart;
#pragma unroll
                for (int row = 0; row < 4; ++row) {
                    const int pbase = ((row + r) * P_COLS + basecol) * IC_PAD + boff;
                    const v8bf* bh = (const v8bf*)(ph + pbase);
                    const v8bf* bl = (const v8bf*)(pl + pbase);
                    v16bf Bh = cat16(bh[0], bh[1]);
                    v16bf Bl = cat16(bl[0], bl[1]);
                    // hi*hi + hi*lo + lo*hi  (lo*lo dropped, ~2^-16 of signal)
                    acc[row] = __builtin_amdgcn_wmma_f32_16x16x32_bf16(
                            false, Ah, false, Bh, (short)0, acc[row], false, false);
                    acc[row] = __builtin_amdgcn_wmma_f32_16x16x32_bf16(
                            false, Ah, false, Bl, (short)0, acc[row], false, false);
                    acc[row] = __builtin_amdgcn_wmma_f32_16x16x32_bf16(
                            false, Al, false, Bh, (short)0, acc[row], false, false);
                }
            }
        }
    }

    // store with static crop mask: zero rows/cols [44,84)
    const bool colMasked = (wp >= 44) && (wp < 84);
#pragma unroll
    for (int row = 0; row < 4; ++row) {
        const int  h    = h0 + row;
        const float mask = ((h >= 44) && (h < 84) && colMasked) ? 0.0f : 1.0f;
        for (int v = 0; v < 8; ++v) {
            int oc = oc0 + half * 8 + v;
            out[((b * OUT_C + oc) * HH + h) * WWID + wp] = acc[row][v] * mask;
        }
    }
}

extern "C" void kernel_launch(void* const* d_in, const int* in_sizes, int n_in,
                              void* d_out, int out_size, void* d_ws, size_t ws_size,
                              hipStream_t stream) {
    const float* x   = (const float*)d_in[0];   // [16,64,128,128] fp32
    const float* wgt = (const float*)d_in[1];   // [64,64,3,3] fp32
    float*       out = (float*)d_out;           // [16,64,128,128] fp32

    bf16_t* ws_hi = (bf16_t*)d_ws;              // 72 KiB
    bf16_t* ws_lo = ws_hi + FRAG_VALS;          // 72 KiB

    // 1) pack weights into WMMA fragment order (hi/lo bf16 splits)
    hipLaunchKernelGGL(pack_w_bf16x2, dim3((FRAG_VALS + 255) / 256), dim3(256),
                       0, stream, wgt, ws_hi, ws_lo);
    // 2) conv + crop mask: b * (128/4 row-groups) * (128/32 strips) blocks
    hipLaunchKernelGGL(cropconv_wmma_bf16x3, dim3(16 * 32 * 4), dim3(256),
                       0, stream, x, ws_hi, ws_lo, out);
}
